// LSTM_4080218931287
// MI455X (gfx1250) — matrix operand order repaired and checked by reading, not compiled, not run on hardware
//
#include <hip/hip_runtime.h>
#include <hip/hip_bf16.h>

// Problem sizes (fixed by the reference)
#define TT  512
#define BB  256
#define DD  512
#define HH  512
#define H2  1024   // 2*H

typedef __attribute__((ext_vector_type(16))) _Float16 v16h;
typedef __attribute__((ext_vector_type(8)))  _Float16 v8h;
typedef __attribute__((ext_vector_type(8)))  float    v8f;
typedef __attribute__((ext_vector_type(4)))  float    v4f;
typedef unsigned int u32x4 __attribute__((ext_vector_type(4)));
typedef int          i32x8 __attribute__((ext_vector_type(8)));
typedef int          i32x4 __attribute__((ext_vector_type(4)));

#if defined(__has_builtin)
#if __has_builtin(__builtin_amdgcn_tensor_load_to_lds) && __has_builtin(__builtin_amdgcn_s_wait_tensorcnt)
#define HAVE_TDM 1
#endif
#endif

// ---------------- helpers ----------------

__device__ __forceinline__ v8f wmma_f16(v16h a, v16h b, v8f c) {
  // D = A(16x32 f16) * B(32x16 f16) + C(16x16 f32)
  return __builtin_amdgcn_wmma_f32_16x16x32_f16(
      /*neg_a=*/false, a, /*neg_b=*/false, b,
      /*c_mod=*/(short)0, c, /*reuse_a=*/false, /*reuse_b=*/false);
}

// A-fragment: two contiguous 8-element f16 runs (K = kb+16*half .. +7 and +8)
__device__ __forceinline__ v16h frag_f16_2x8(const _Float16* p0, const _Float16* p1) {
  v8h lo = *(const v8h*)p0;
  v8h hi = *(const v8h*)p1;
  v16h r;
#pragma unroll
  for (int i = 0; i < 8; ++i) { r[i] = lo[i]; r[8 + i] = hi[i]; }
  return r;
}

// B-fragment: one contiguous 16-element f16 run (K = kb+16*half .. +15)
__device__ __forceinline__ v16h frag_f16_16(const _Float16* p) {
  return frag_f16_2x8(p, p + 8);
}

__device__ __forceinline__ float sigmoid_f(float x) {
  return 1.0f / (1.0f + __expf(-x));
}
__device__ __forceinline__ float tanh_f(float x) {
  float e = __expf(-2.0f * x);
  return (1.0f - e) / (1.0f + e);
}

#ifdef HAVE_TDM
// 1-D Tensor-DMA copy of n fp32 elements (contiguous) global -> LDS.
// Issued once per wave that executes it (TDM ignores EXEC). D# per ISA 08_async_tensor.
// This toolchain's builtin arity: (u32x4 g0, i32x8 g1, i32x4 g2, i32x4 g3, i32x8 g4, i32 cpol)
__device__ __forceinline__ void tdm_load_f32_1d(const float* gsrc, float* lds_dst, unsigned n) {
  unsigned long long ga = (unsigned long long)(uintptr_t)gsrc;
  unsigned lds = (unsigned)(uintptr_t)lds_dst;   // low 32 bits = LDS offset
  u32x4 g0;
  g0[0] = 1u;                                     // count=1, user D#
  g0[1] = lds;                                    // lds_addr
  g0[2] = (unsigned)(ga & 0xFFFFFFFFu);           // global_addr[31:0]
  g0[3] = (unsigned)((ga >> 32) & 0x1FFFFFFu) | (2u << 30);  // global_addr[56:32] | type=2
  i32x8 g1;
  g1[0] = (int)(2u << 16);                        // data_size = 4B
  g1[1] = (int)((n & 0xFFFFu) << 16);             // tensor_dim0[15:0]  (bits 63:48)
  g1[2] = (int)((n >> 16) & 0xFFFFu) | (int)(1u << 16); // tensor_dim0 hi | tensor_dim1 = 1
  g1[3] = (int)((n & 0xFFFFu) << 16);             // tile_dim0 (bits 127:112)
  g1[4] = 1;                                      // tile_dim1 = 1, tile_dim2 = 0
  g1[5] = (int)n;                                 // tensor_dim0_stride low 32
  g1[6] = 0;
  g1[7] = 0;
  i32x4 z4; z4[0] = 0; z4[1] = 0; z4[2] = 0; z4[3] = 0;
  i32x8 z8;
#pragma unroll
  for (int i = 0; i < 8; ++i) z8[i] = 0;
  __builtin_amdgcn_tensor_load_to_lds(g0, g1, z4, z4, z8, 0);
}
#endif

// ---------------- kernel 0: fp32 -> f16 weight conversion ----------------

__global__ void cvt_f32_f16_kernel(const float* __restrict__ src,
                                   _Float16* __restrict__ dst, int n) {
  int i = blockIdx.x * blockDim.x + threadIdx.x;
  if (i < n) dst[i] = (_Float16)src[i];
}

// ---------------- kernel 1: G = input @ Wi^T + bi  (parallel over T*B) ----
// M = T*B = 131072, N = 2H = 1024, K = D = 512.
// One 16-row M-tile per workgroup, staged via Tensor Data Mover into LDS,
// converted once to f16; 8 waves each own a 128-wide N strip with
// register-double-buffered B fragments so VMEM overlaps WMMA.

__global__ void __launch_bounds__(256)
input_proj_kernel(const float* __restrict__ x,       // (T*B, D) fp32
                  const _Float16* __restrict__ Wih,  // (2H, D) f16 row-major
                  const float* __restrict__ bi,      // (2H)
                  float* __restrict__ G)             // (T*B, 2H) fp32
{
  __shared__ float    xstage[16 * DD];   // 32 KB fp32 staging (TDM target)
  __shared__ _Float16 xh[16 * DD];       // 16 KB f16 A-tile

  const int mtile = blockIdx.x;           // 16 rows each
  const int wave  = threadIdx.x >> 5;     // 0..7
  const int lane  = threadIdx.x & 31;
  const int half  = lane >> 4;            // 0|1
  const int idx   = lane & 15;

#ifdef HAVE_TDM
  if (threadIdx.x < 32) {                 // wave 0 issues the DMA (one instr per wave)
    tdm_load_f32_1d(x + (size_t)mtile * 16 * DD, xstage, 16 * DD);
    __builtin_amdgcn_s_wait_tensorcnt(0);
  }
#else
  for (int i = threadIdx.x; i < 16 * DD; i += 256)
    xstage[i] = x[(size_t)mtile * 16 * DD + i];
#endif
  __syncthreads();
  for (int i = threadIdx.x; i < 16 * DD; i += 256)
    xh[i] = (_Float16)xstage[i];
  __syncthreads();

  const int nbase = wave * 128;
  const _Float16* wbase = Wih + (size_t)(nbase + idx) * DD;

  v8f acc[8];
#pragma unroll
  for (int q = 0; q < 8; ++q)
#pragma unroll
    for (int j = 0; j < 8; ++j) acc[q][j] = 0.0f;

  // double-buffered B fragments
  v16h bcur[8], bnxt[8];
#pragma unroll
  for (int q = 0; q < 8; ++q)
    bcur[q] = frag_f16_16(wbase + (size_t)q * 16 * DD + half * 16);

  for (int kb = 0; kb < DD; kb += 32) {
    const int ka0 = kb + half * 16;
    const int ka1 = kb + half * 16 + 8;
    v16h afrag = frag_f16_2x8(&xh[idx * DD + ka0], &xh[idx * DD + ka1]);
    const int kn = kb + 32;
    if (kn < DD) {
      const int kbB = kn + half * 16;
#pragma unroll
      for (int q = 0; q < 8; ++q)
        bnxt[q] = frag_f16_16(wbase + (size_t)q * 16 * DD + kbB);
    }
#pragma unroll
    for (int q = 0; q < 8; ++q)
      acc[q] = wmma_f16(afrag, bcur[q], acc[q]);
#pragma unroll
    for (int q = 0; q < 8; ++q) bcur[q] = bnxt[q];
  }

#pragma unroll
  for (int q = 0; q < 8; ++q) {
    const int n = nbase + q * 16 + idx;
    const float bias = bi[n];
#pragma unroll
    for (int j = 0; j < 8; ++j) {
      const int r = mtile * 16 + j + 8 * half;
      G[(size_t)r * H2 + n] = acc[q][j] + bias;
    }
  }
}

// ---------------- kernel 2: persistent recurrence ----------------
// grid = B/16 = 16 workgroups, 512 threads = 16 waves (4 per SIMD32 for
// latency hiding). Each WG owns 16 batch rows; wave w owns H-columns
// [w*32, w*32+32). State: v,a fp32 in registers (C-layout); f16 LDS shadows
// feed the WMMA A-operands. G tile for step t is preloaded into registers at
// the top of P1 (overlaps the GEMM), and step t+1's tile is prefetched.

__global__ void __launch_bounds__(512)
recurrent_kernel(const float* __restrict__ G,        // (T*B, 2H) fp32
                 const _Float16* __restrict__ Wvh,   // (2H, H) f16
                 const float* __restrict__ bv,       // (2H)
                 const _Float16* __restrict__ Wah,   // (H, H) f16
                 const float* __restrict__ ba,       // (H)
                 const _Float16* __restrict__ Wvah,  // (H, H) f16
                 const float* __restrict__ bva,      // (H)
                 float* __restrict__ out)            // T*B*H | v_fin B*H | a_fin B*H
{
  __shared__ _Float16 vh[16 * HH];    // 16 KB  f16 shadow of v tile
  __shared__ _Float16 ah[16 * HH];    // 16 KB  f16 shadow of a tile
  __shared__ _Float16 dvh[16 * HH];   // 16 KB  f16 shadow of dv tile

  const int wg   = blockIdx.x;        // batch tile (16 rows)
  const int wave = threadIdx.x >> 5;  // 0..15
  const int lane = threadIdx.x & 31;
  const int half = lane >> 4;
  const int idx  = lane & 15;

  for (int i = threadIdx.x; i < 16 * HH; i += blockDim.x) {
    vh[i] = (_Float16)0.0f;
    ah[i] = (_Float16)0.0f;
    dvh[i] = (_Float16)0.0f;
  }

  // persistent fp32 state in registers, C-layout: elem (m=j+8*half, n=wave*32+q*16+idx)
  v8f vreg[2], areg[2];
#pragma unroll
  for (int q = 0; q < 2; ++q)
#pragma unroll
    for (int j = 0; j < 8; ++j) { vreg[q][j] = 0.0f; areg[q][j] = 0.0f; }

  float bvlo[2], bvhi[2], bar[2], bvar[2];
#pragma unroll
  for (int q = 0; q < 2; ++q) {
    const int n = wave * 32 + q * 16 + idx;
    bvlo[q] = bv[n];
    bvhi[q] = bv[HH + n];
    bar[q]  = ba[n];
    bvar[q] = bva[n];
  }

  __syncthreads();

  for (int t = 0; t < TT; ++t) {
    const float* Gt = G + ((size_t)t * BB + (size_t)wg * 16) * H2;

    // ---- P1: preload G(t) into registers, then the three GEMMs ----
    float gl[2][8], gh[2][8];
#pragma unroll
    for (int q = 0; q < 2; ++q) {
      const int n = wave * 32 + q * 16 + idx;
#pragma unroll
      for (int j = 0; j < 8; ++j) {
        const int m = j + 8 * half;
        gl[q][j] = Gt[(size_t)m * H2 + n];
        gh[q][j] = Gt[(size_t)m * H2 + HH + n];
      }
    }

    v8f accLo[2], accHi[2], accA[2];
#pragma unroll
    for (int q = 0; q < 2; ++q)
#pragma unroll
      for (int j = 0; j < 8; ++j) { accLo[q][j] = 0.0f; accHi[q][j] = 0.0f; accA[q][j] = 0.0f; }

    for (int kb = 0; kb < HH; kb += 32) {
      const int ka0 = kb + half * 16;
      const int ka1 = kb + half * 16 + 8;
      const int kbB = kb + half * 16;
      v16h av = frag_f16_2x8(&vh[idx * HH + ka0], &vh[idx * HH + ka1]);
      v16h aa = frag_f16_2x8(&ah[idx * HH + ka0], &ah[idx * HH + ka1]);
#pragma unroll
      for (int q = 0; q < 2; ++q) {
        const int n = wave * 32 + q * 16 + idx;
        v16h blo = frag_f16_16(Wvh + (size_t)n * HH + kbB);
        accLo[q] = wmma_f16(av, blo, accLo[q]);
        v16h bhi = frag_f16_16(Wvh + (size_t)(HH + n) * HH + kbB);
        accHi[q] = wmma_f16(av, bhi, accHi[q]);
        v16h bwa = frag_f16_16(Wah + (size_t)n * HH + kbB);
        accA[q] = wmma_f16(aa, bwa, accA[q]);
      }
    }
    __syncthreads();   // all waves done reading vh/ah

    // ---- P2: gates, dv, v_new; write vh/dvh shadows + output row ----
    v8f f_r[2], dv_r[2];
#pragma unroll
    for (int q = 0; q < 2; ++q) {
      const int n = wave * 32 + q * 16 + idx;
#pragma unroll
      for (int j = 0; j < 8; ++j) {
        const int m = j + 8 * half;
        const float ag  = sigmoid_f(gl[q][j] + bvlo[q] + accLo[q][j]);
        const float fg  = sigmoid_f(gh[q][j] + bvhi[q] + accHi[q][j]);
        const float a2v = tanh_f(accA[q][j] + bar[q]);
        const float dv  = a2v * ag;
        const float vnw = vreg[q][j] + dv;
        vreg[q][j] = vnw;
        f_r[q][j]  = fg;
        dv_r[q][j] = dv;
        vh[m * HH + n]  = (_Float16)vnw;
        dvh[m * HH + n] = (_Float16)dv;
        out[(((size_t)t * BB) + wg * 16 + m) * HH + n] = vnw;
      }
    }
    __syncthreads();   // dvh fully written

    // prefetch next step's 64KB G tile (512 threads x 128B)
    if (t + 1 < TT) {
      const char* Gn = (const char*)(G + ((size_t)(t + 1) * BB + (size_t)wg * 16) * H2);
      __builtin_prefetch(Gn + (size_t)threadIdx.x * 128, 0, 0);
    }

    // ---- P3: accV = dv @ Wva^T ----
    v8f accV[2];
#pragma unroll
    for (int q = 0; q < 2; ++q)
#pragma unroll
      for (int j = 0; j < 8; ++j) accV[q][j] = 0.0f;

    for (int kb = 0; kb < HH; kb += 32) {
      const int ka0 = kb + half * 16;
      const int ka1 = kb + half * 16 + 8;
      const int kbB = kb + half * 16;
      v16h ad = frag_f16_2x8(&dvh[idx * HH + ka0], &dvh[idx * HH + ka1]);
#pragma unroll
      for (int q = 0; q < 2; ++q) {
        const int n = wave * 32 + q * 16 + idx;
        v16h bw = frag_f16_16(Wvah + (size_t)n * HH + kbB);
        accV[q] = wmma_f16(ad, bw, accV[q]);
      }
    }

    // ---- P4: a_new = a*f + tanh(dv)*sigmoid(dv@Wva^T + bva); update ah ----
#pragma unroll
    for (int q = 0; q < 2; ++q) {
      const int n = wave * 32 + q * 16 + idx;
#pragma unroll
      for (int j = 0; j < 8; ++j) {
        const int m = j + 8 * half;
        const float nag = sigmoid_f(accV[q][j] + bvar[q]);
        const float anw = areg[q][j] * f_r[q][j] + tanh_f(dv_r[q][j]) * nag;
        areg[q][j] = anw;
        ah[m * HH + n] = (_Float16)anw;
      }
    }
    __syncthreads();   // ah/vh stable before next step's P1 reads
  }

  // ---- final states ----
  float* vf = out + (size_t)TT * BB * HH;
  float* af = vf + (size_t)BB * HH;
#pragma unroll
  for (int q = 0; q < 2; ++q) {
    const int n = wave * 32 + q * 16 + idx;
#pragma unroll
    for (int j = 0; j < 8; ++j) {
      const int m = j + 8 * half;
      vf[(size_t)(wg * 16 + m) * HH + n] = vreg[q][j];
      af[(size_t)(wg * 16 + m) * HH + n] = areg[q][j];
    }
  }
}

// ---------------- host launch ----------------

extern "C" void kernel_launch(void* const* d_in, const int* in_sizes, int n_in,
                              void* d_out, int out_size, void* d_ws, size_t ws_size,
                              hipStream_t stream) {
  (void)in_sizes; (void)n_in; (void)out_size; (void)ws_size;

  const float* input = (const float*)d_in[0];  // (T, B, D)
  const float* Wi    = (const float*)d_in[1];  // (2H, D)
  const float* bi    = (const float*)d_in[2];  // (2H)
  const float* Wv    = (const float*)d_in[3];  // (2H, H)
  const float* bv    = (const float*)d_in[4];  // (2H)
  const float* Wa    = (const float*)d_in[5];  // (H, H)
  const float* ba    = (const float*)d_in[6];  // (H)
  const float* Wva   = (const float*)d_in[7];  // (H, H)
  const float* bva   = (const float*)d_in[8];  // (H)
  float* out = (float*)d_out;

  // workspace layout (bytes):
  //   [0, 1MB)         Wi  f16  (2H*D)
  //   [1MB, 2MB)       Wv  f16  (2H*H)
  //   [2MB, 2.5MB)     Wa  f16  (H*H)
  //   [2.5MB, 3MB)     Wva f16  (H*H)
  //   [3MB, 3MB+512MB) G   fp32 (T*B, 2H)
  char* ws = (char*)d_ws;
  _Float16* Wih  = (_Float16*)ws;
  _Float16* Wvh  = Wih + (size_t)H2 * DD;
  _Float16* Wah  = Wvh + (size_t)H2 * HH;
  _Float16* Wvah = Wah + (size_t)HH * HH;
  float* G = (float*)(ws + (size_t)3 * 1024 * 1024);

  const int cvtThreads = 256;
  cvt_f32_f16_kernel<<<(H2 * DD + cvtThreads - 1) / cvtThreads, cvtThreads, 0, stream>>>(Wi, Wih, H2 * DD);
  cvt_f32_f16_kernel<<<(H2 * HH + cvtThreads - 1) / cvtThreads, cvtThreads, 0, stream>>>(Wv, Wvh, H2 * HH);
  cvt_f32_f16_kernel<<<(HH * HH + cvtThreads - 1) / cvtThreads, cvtThreads, 0, stream>>>(Wa, Wah, HH * HH);
  cvt_f32_f16_kernel<<<(HH * HH + cvtThreads - 1) / cvtThreads, cvtThreads, 0, stream>>>(Wva, Wvah, HH * HH);

  // Big parallel GEMM: G = input @ Wi^T + bi  (TDM-staged A tiles)
  input_proj_kernel<<<(TT * BB) / 16, 256, 0, stream>>>(input, Wih, bi, G);

  // Persistent recurrence: 16 WGs (one 16-row batch tile each), 16 waves each
  recurrent_kernel<<<BB / 16, 512, 0, stream>>>(G, Wvh, bv, Wah, ba, Wvah, bva, out);
}
